// CrossAttentionFusion_19748259627539
// MI455X (gfx1250) — matrix operand order repaired and checked by reading, not compiled, not run on hardware
//
#include <hip/hip_runtime.h>
#include <math.h>

// ---------------------------------------------------------------------------
// CrossAttentionFusion for MI455X (gfx1250): flash-attention with f16 WMMA,
// TDM-staged V tiles, exp2-domain online softmax, software-pipelined
// LDS fragment fetch.
// B=8, C=256, H=W=64 -> N=4096, CQK=32.
// ---------------------------------------------------------------------------

#define BATCH 8
#define CCH   256
#define NSEQ  4096
#define CQKD  32

typedef __attribute__((ext_vector_type(16))) _Float16     v16h;
typedef __attribute__((ext_vector_type(8)))  _Float16     v8h;
typedef __attribute__((ext_vector_type(8)))  float        v8f;
typedef __attribute__((ext_vector_type(4)))  unsigned int u32x4;
typedef __attribute__((ext_vector_type(8)))  int          i32x8;
typedef __attribute__((ext_vector_type(4)))  int          i32x4;

// TDM builtin: 5-arg form probe-confirmed on ROCm 7.2 (clang-22) only.
#if defined(__HIP_DEVICE_COMPILE__) && __has_builtin(__builtin_amdgcn_tensor_load_to_lds) && (__clang_major__ < 23)
#define USE_TDM 1
#else
#define USE_TDM 0
#endif

// Build a 16-half A/B fragment from two contiguous 8-half (16B) chunks.
__device__ __forceinline__ v16h make_frag(const _Float16* p0, const _Float16* p1) {
    v8h lo = *(const v8h*)p0;
    v8h hi = *(const v8h*)p1;
    v16h r;
#pragma unroll
    for (int t = 0; t < 8; ++t) { r[t] = lo[t]; r[t + 8] = hi[t]; }
    return r;
}

// A-fragment from an f32 row (weight matrix), converting to f16.
__device__ __forceinline__ v16h load_a_f32(const float* row, int kb) {
    v16h r;
#pragma unroll
    for (int t = 0; t < 8; ++t) {
        r[t]     = (_Float16)row[2 * kb + t];
        r[t + 8] = (_Float16)row[2 * kb + 8 + t];
    }
    return r;
}

__device__ __forceinline__ v8f wmma_f16(v16h a, v16h b, v8f c) {
    return __builtin_amdgcn_wmma_f32_16x16x32_f16(false, a, false, b,
                                                  (short)0, c, false, false);
}

// Hardware base-2 exponential (v_exp_f32).
__device__ __forceinline__ float exp2_hw(float x) {
    return __builtin_amdgcn_exp2f(x);
}

// XOR shuffle via ds_swizzle (group-of-32: offset = xor<<10 | and 0x1F).
// Masks 1/2/4/8 stay inside each 16-lane half = one WMMA C/D row group.
template <int MASK>
__device__ __forceinline__ float swz(float v) {
    int i = __builtin_amdgcn_ds_swizzle(__float_as_int(v), (MASK << 10) | 0x1F);
    return __int_as_float(i);
}
__device__ __forceinline__ float rowmax16(float v) {
    v = fmaxf(v, swz<1>(v)); v = fmaxf(v, swz<2>(v));
    v = fmaxf(v, swz<4>(v)); v = fmaxf(v, swz<8>(v));
    return v;
}
__device__ __forceinline__ float rowsum16(float v) {
    v += swz<1>(v); v += swz<2>(v); v += swz<4>(v); v += swz<8>(v);
    return v;
}

// ---------------------------------------------------------------------------
// Projection GEMM:  Out[m, n] = (sum_k W[m,k] * X[k, n0+n] + bias[m]) * scale
// ---------------------------------------------------------------------------
template <bool IN_F16, bool OUT_F16, bool TRANS_OUT, bool RESID>
__global__ __launch_bounds__(256) void proj_kernel(
    const void* __restrict__ Xin, const float* __restrict__ W,
    const float* __restrict__ bias, const float* __restrict__ resid,
    void* __restrict__ Out, int M, float scale)
{
    __shared__ _Float16 Xs[64][264];   // [n][k], padded row (528B, 16B-aligned)

    const int tid  = threadIdx.x;
    const int wave = tid >> 5;
    const int lane = tid & 31;
    const int nl   = lane & 15;
    const int half = lane >> 4;
    const int kb   = half * 8;

    const int b  = blockIdx.x >> 6;
    const int n0 = (blockIdx.x & 63) * 64;

    if constexpr (IN_F16) {
        const _Float16* X = (const _Float16*)Xin + (size_t)b * CCH * NSEQ;
        for (int idx = tid; idx < 256 * 64; idx += 256) {
            int k = idx >> 6, n = idx & 63;
            Xs[n][k] = X[(size_t)k * NSEQ + n0 + n];
        }
    } else {
        const float* X = (const float*)Xin + (size_t)b * CCH * NSEQ;
        for (int idx = tid; idx < 256 * 64; idx += 256) {
            int k = idx >> 6, n = idx & 63;
            Xs[n][k] = (_Float16)X[(size_t)k * NSEQ + n0 + n];
        }
    }
    __syncthreads();

    const int numTiles = (M >> 4) * 4;
    for (int t = wave; t < numTiles; t += 8) {
        const int mt = t >> 2, nt = t & 3;
        v8f acc = {0.f, 0.f, 0.f, 0.f, 0.f, 0.f, 0.f, 0.f};
        const float*    wrow = W + (size_t)(mt * 16 + nl) * CCH;
        const _Float16* xrow = &Xs[nt * 16 + nl][half * 16];
        // one-stage software pipeline over the K loop
        v16h aF = load_a_f32(wrow, kb);
        v16h bF = make_frag(xrow, xrow + 8);
#pragma unroll
        for (int k0 = 0; k0 < CCH; k0 += 32) {
            v16h aN = aF, bN = bF;
            if (k0 + 32 < CCH) {
                aN = load_a_f32(wrow + k0 + 32, kb);
                bN = make_frag(xrow + k0 + 32, xrow + k0 + 40);
            }
            acc = wmma_f16(aF, bF, acc);
            aF = aN; bF = bN;
        }
        const int ncol = n0 + nt * 16 + nl;
        if constexpr (TRANS_OUT) {
            _Float16* OT = (_Float16*)Out + (size_t)b * NSEQ * M;
            v8h pack;
#pragma unroll
            for (int r = 0; r < 8; ++r) {
                int m = mt * 16 + 8 * half + r;
                pack[r] = (_Float16)((acc[r] + bias[m]) * scale);
            }
            *(v8h*)(OT + (size_t)ncol * M + mt * 16 + 8 * half) = pack;
        } else if constexpr (OUT_F16) {
            _Float16* O = (_Float16*)Out + (size_t)b * M * NSEQ;
#pragma unroll
            for (int r = 0; r < 8; ++r) {
                int m = mt * 16 + 8 * half + r;
                O[(size_t)m * NSEQ + ncol] = (_Float16)((acc[r] + bias[m]) * scale);
            }
        } else {
            float* O = (float*)Out + (size_t)b * M * NSEQ;
            const float* R = resid + (size_t)b * M * NSEQ;
#pragma unroll
            for (int r = 0; r < 8; ++r) {
                int m = mt * 16 + 8 * half + r;
                size_t off = (size_t)m * NSEQ + ncol;
                float v = acc[r] + bias[m];
                if constexpr (RESID) v += R[off];
                O[off] = v;
            }
        }
    }
}

// ---------------------------------------------------------------------------
// Flash attention: 128 threads (4 waves), 64 query rows per block, 16 per
// wave; 64-key tiles. S via 4 WMMAs (K=CQK=32), exp2-domain online softmax
// with ds_swizzle row reductions + conditional accumulator rescale,
// O += P*V via 32 WMMAs with one-tile-ahead B-fragment prefetch. V tile
// staged by TDM (tensor_load_to_lds) with LDS padding to 72-halfword rows.
// qT/kT: [B][N][32] f16 (q pre-scaled by log2e/sqrt(32)). vS: [B][256][N] f16.
// ---------------------------------------------------------------------------
__global__ __launch_bounds__(128) void attn_kernel(
    const _Float16* __restrict__ qT, const _Float16* __restrict__ kT,
    const _Float16* __restrict__ vS, _Float16* __restrict__ aOut)
{
    __shared__ _Float16 Ks[64][40];        // key tile  [j][d]          5.0 KB
    __shared__ _Float16 Vs[256][72];       // value tile [c][j] padded 36.0 KB
    __shared__ _Float16 Ps[4][16][72];     // per-wave P tile [i][j]    9.0 KB

    const int tid  = threadIdx.x;
    const int wave = tid >> 5;
    const int lane = tid & 31;
    const int nl   = lane & 15;
    const int half = lane >> 4;
    const int kb   = half * 8;

    const int b  = blockIdx.x >> 6;
    const int i0 = (blockIdx.x & 63) * 64 + wave * 16;

    const _Float16* qTb = qT + (size_t)b * NSEQ * CQKD;
    const _Float16* kTb = kT + (size_t)b * NSEQ * CQKD;
    const _Float16* vSb = vS + (size_t)b * CCH * NSEQ;

    const _Float16* qrow = qTb + (size_t)(i0 + nl) * CQKD;
    const v16h qA = make_frag(qrow + 2 * kb, qrow + 2 * kb + 8);

    v8f acc[16];
#pragma unroll
    for (int ct = 0; ct < 16; ++ct)
#pragma unroll
        for (int r = 0; r < 8; ++r) acc[ct][r] = 0.f;

    float mrow[8], lrow[8];
#pragma unroll
    for (int r = 0; r < 8; ++r) { mrow[r] = -3.0e38f; lrow[r] = 0.f; }

    const int ch4 = tid & 3, rw4 = tid >> 2;   // K-tile loaders
#if !USE_TDM
    const int ch8 = tid & 7, rw8 = tid >> 3;   // V-tile loaders (fallback)
#endif

    for (int j0 = 0; j0 < NSEQ; j0 += 64) {
        // ---- stage K tile (64x32 f16, 16B chunks) -------------------------
#pragma unroll
        for (int u = 0; u < 2; ++u) {
            int j = rw4 + u * 32;
            *(v8h*)&Ks[j][ch4 * 8] =
                *(const v8h*)(kTb + (size_t)(j0 + j) * CQKD + ch4 * 8);
        }

        // ---- stage V tile (256x64 f16) ------------------------------------
#if USE_TDM
        if (wave == 0) {
            // D# group 0: count=1 | lds_addr | global_addr(57b) | type=2
            unsigned long long ga =
                (unsigned long long)(uintptr_t)vSb + (unsigned)(j0 * 2);
            u32x4 g0 = { 1u,
                         (unsigned)(uintptr_t)&Vs[0][0],
                         (unsigned)(ga & 0xFFFFFFFFull),
                         (unsigned)((ga >> 32) & 0x1FFFFFFull) | (2u << 30) };
            // D# group 1: data_size=2B, pad_enable, pad every 32 DWORDs
            // (=128B tile row) by 4 DWORDs (-> 144B LDS row stride).
            // tensor_dim0=4096, tensor_dim1=256, tile_dim0=64, tile_dim1=256,
            // tensor_dim0_stride=4096.
            i32x8 g1 = { (1 << 16) | (1 << 20) | (4 << 22) | (3 << 25),
                         (int)(4096u << 16),
                         (int)(256u << 16),
                         (int)(64u << 16),
                         256, 4096, 0, 0 };
            i32x4 gz = { 0, 0, 0, 0 };
            __builtin_amdgcn_tensor_load_to_lds(g0, g1, gz, gz, 0);
            __builtin_amdgcn_s_wait_tensorcnt(0);
        }
#else
#pragma unroll
        for (int u = 0; u < 16; ++u) {
            int c = rw8 + u * 16;
            *(v8h*)&Vs[c][ch8 * 8] =
                *(const v8h*)(vSb + (size_t)c * NSEQ + j0 + ch8 * 8);
        }
        if (j0 + 64 < NSEQ)
            __builtin_prefetch(vSb + (size_t)(tid * 2) * NSEQ + j0 + 64, 0, 1);
#endif
        if (j0 + 64 < NSEQ)
            __builtin_prefetch(kTb + (size_t)(j0 + 64 + rw4) * CQKD, 0, 1);
        __syncthreads();

        // ---- S = q . k^T: fetch all 4 K-fragments, then 4 WMMAs -----------
        v8f z = {0.f, 0.f, 0.f, 0.f, 0.f, 0.f, 0.f, 0.f};
        v16h kf[4];
#pragma unroll
        for (int jt = 0; jt < 4; ++jt) {
            const _Float16* kr = &Ks[jt * 16 + nl][half * 16];
            kf[jt] = make_frag(kr, kr + 8);
        }
        v8f s[4];
#pragma unroll
        for (int jt = 0; jt < 4; ++jt) s[jt] = wmma_f16(qA, kf[jt], z);

        // ---- exp2-domain online softmax -----------------------------------
        float p[4][8], sc[8];
        bool changed = false;
#pragma unroll
        for (int r = 0; r < 8; ++r) {
            float tmax = fmaxf(fmaxf(s[0][r], s[1][r]), fmaxf(s[2][r], s[3][r]));
            tmax = rowmax16(tmax);
            float mnew = fmaxf(mrow[r], tmax);
            sc[r] = exp2_hw(mrow[r] - mnew);
            changed |= (mnew != mrow[r]);
            float psum = 0.f;
#pragma unroll
            for (int jt = 0; jt < 4; ++jt) {
                p[jt][r] = exp2_hw(s[jt][r] - mnew);
                psum += p[jt][r];
            }
            lrow[r] = lrow[r] * sc[r] + rowsum16(psum);
            mrow[r] = mnew;
        }
        if (__ballot(changed)) {           // uniform skip when max unchanged
#pragma unroll
            for (int ct = 0; ct < 16; ++ct)
#pragma unroll
                for (int r = 0; r < 8; ++r) acc[ct][r] *= sc[r];
        }

        // ---- D-layout -> A-layout for P via per-wave LDS round-trip -------
#pragma unroll
        for (int r = 0; r < 8; ++r) {
            int row = r + 8 * half;
#pragma unroll
            for (int jt = 0; jt < 4; ++jt)
                Ps[wave][row][jt * 16 + nl] = (_Float16)p[jt][r];
        }
        asm volatile("s_wait_dscnt 0" ::: "memory");   // in-order LDS RAW
        v16h pA0 = make_frag(&Ps[wave][nl][2 * kb],      &Ps[wave][nl][2 * kb + 8]);
        v16h pA1 = make_frag(&Ps[wave][nl][32 + 2 * kb], &Ps[wave][nl][40 + 2 * kb]);

        // ---- O[i, c] += P[i, j] * V[c, j], B-frags fetched 1 tile ahead ---
        {
            const _Float16* vr0 = &Vs[nl][half * 16];
            v16h vb0 = make_frag(vr0, vr0 + 8);
            v16h vb1 = make_frag(vr0 + 32, vr0 + 40);
#pragma unroll
            for (int ct = 0; ct < 16; ++ct) {
                v16h nb0 = vb0, nb1 = vb1;
                if (ct + 1 < 16) {
                    const _Float16* vr = &Vs[(ct + 1) * 16 + nl][half * 16];
                    nb0 = make_frag(vr, vr + 8);
                    nb1 = make_frag(vr + 32, vr + 40);
                }
                acc[ct] = wmma_f16(pA0, vb0, acc[ct]);
                acc[ct] = wmma_f16(pA1, vb1, acc[ct]);
                vb0 = nb0; vb1 = nb1;
            }
        }
        __syncthreads();
    }

    // ---- normalize + store: aOut[c][i], 8 rows -> one 16B store -----------
    float linv[8];
#pragma unroll
    for (int r = 0; r < 8; ++r) linv[r] = 1.f / lrow[r];
    _Float16* outb = aOut + (size_t)b * CCH * NSEQ;
#pragma unroll
    for (int ct = 0; ct < 16; ++ct) {
        int c = ct * 16 + nl;
        v8h pack;
#pragma unroll
        for (int r = 0; r < 8; ++r) pack[r] = (_Float16)(acc[ct][r] * linv[r]);
        *(v8h*)(outb + (size_t)c * NSEQ + i0 + 8 * half) = pack;
    }
}

// ---------------------------------------------------------------------------
extern "C" void kernel_launch(void* const* d_in, const int* in_sizes, int n_in,
                              void* d_out, int out_size, void* d_ws, size_t ws_size,
                              hipStream_t stream) {
    const float* yolo = (const float*)d_in[0];
    const float* vit  = (const float*)d_in[1];
    const float* Wq   = (const float*)d_in[2];
    const float* bq   = (const float*)d_in[3];
    const float* Wk   = (const float*)d_in[4];
    const float* bk   = (const float*)d_in[5];
    const float* Wv   = (const float*)d_in[6];
    const float* bv   = (const float*)d_in[7];
    const float* Wo   = (const float*)d_in[8];
    const float* bo   = (const float*)d_in[9];

    _Float16* qT = (_Float16*)d_ws;
    _Float16* kT = qT + (size_t)BATCH * NSEQ * CQKD;
    _Float16* vS = kT + (size_t)BATCH * NSEQ * CQKD;
    _Float16* aO = vS + (size_t)BATCH * CCH * NSEQ;

    const int nBlocks = BATCH * (NSEQ / 64);              // 512
    // q pre-scale folds softmax scale AND log2(e) for exp2-domain softmax.
    const float qscale = 0.17677669529663687f * 1.4426950408889634f;

    proj_kernel<false, true, true,  false><<<nBlocks, 256, 0, stream>>>(
        yolo, Wq, bq, nullptr, qT, CQKD, qscale);
    proj_kernel<false, true, true,  false><<<nBlocks, 256, 0, stream>>>(
        vit,  Wk, bk, nullptr, kT, CQKD, 1.0f);
    proj_kernel<false, true, false, false><<<nBlocks, 256, 0, stream>>>(
        vit,  Wv, bv, nullptr, vS, CCH,  1.0f);

    attn_kernel<<<nBlocks, 128, 0, stream>>>(qT, kT, vS, aO);

    proj_kernel<true, false, false, true><<<nBlocks, 256, 0, stream>>>(
        aO, Wo, bo, yolo, d_out, CCH, 1.0f);
}
